// MoELayer_60885456388527
// MI455X (gfx1250) — matrix operand order repaired and checked by reading, not compiled, not run on hardware
//
#include <hip/hip_runtime.h>

#define B_      4
#define S_      2048
#define N_TOK   8192      // B*S
#define DDIM    2048
#define HDIM    4096
#define NEXP    8
#define TILE_M  128
#define TILE_N  128
#define KSTEP   32
#define NKITER  (DDIM / KSTEP)
#define LDSS    40        // halfs per LDS row (80B stride, 16B aligned, conflict-spread)
#define MAXTM   64        // worst case: one expert gets all N tokens -> 8192/128

typedef __attribute__((ext_vector_type(16))) __bf16 v16bf;
typedef __attribute__((ext_vector_type(8)))  float  v8f;

union FragU  { uint4 q[2]; v16bf v; };
union Pack16 { __bf16 b[16]; uint4 q[2]; };

// ---------------------------------------------------------------------------
// Gating: one wave32 per token. gate_w staged in LDS (8x2048 f32 = 64KB).
// ---------------------------------------------------------------------------
__global__ __launch_bounds__(256) void moe_gate(
    const float* __restrict__ x, const float* __restrict__ gate_w,
    const float* __restrict__ gate_b,
    int* __restrict__ cnt, float* __restrict__ ent_acc,
    int* __restrict__ tok_list, float* __restrict__ prob_list)
{
    __shared__ float gws[NEXP * DDIM];
    for (int i = threadIdx.x; i < NEXP * DDIM / 4; i += 256)
        ((float4*)gws)[i] = ((const float4*)gate_w)[i];
    __syncthreads();

    const int lane = threadIdx.x & 31;
    const int wave = threadIdx.x >> 5;
    const int n    = blockIdx.x * 8 + wave;

    const float* xr = x + (long)n * DDIM;
    float part[NEXP];
#pragma unroll
    for (int e = 0; e < NEXP; ++e) part[e] = 0.0f;

    for (int it = 0; it < DDIM / 128; ++it) {
        const int d = (it * 32 + lane) * 4;
        float4 xv = *(const float4*)(xr + d);
#pragma unroll
        for (int e = 0; e < NEXP; ++e) {
            float4 wv = *(const float4*)(&gws[e * DDIM + d]);
            part[e] += xv.x * wv.x + xv.y * wv.y + xv.z * wv.z + xv.w * wv.w;
        }
    }

    float logits[NEXP];
#pragma unroll
    for (int e = 0; e < NEXP; ++e) {
        float v = part[e];
        v += __shfl_xor(v, 16, 32);
        v += __shfl_xor(v,  8, 32);
        v += __shfl_xor(v,  4, 32);
        v += __shfl_xor(v,  2, 32);
        v += __shfl_xor(v,  1, 32);
        logits[e] = v + gate_b[e];
    }

    float m = logits[0];
#pragma unroll
    for (int e = 1; e < NEXP; ++e) m = fmaxf(m, logits[e]);
    float p[NEXP], s = 0.0f;
#pragma unroll
    for (int e = 0; e < NEXP; ++e) { p[e] = __expf(logits[e] - m); s += p[e]; }
    const float inv = 1.0f / s;
    float ent = 0.0f;
#pragma unroll
    for (int e = 0; e < NEXP; ++e) { p[e] *= inv; ent -= p[e] * __logf(p[e] + 1e-10f); }

    // top-2 (lowest index wins ties, matching lax.top_k)
    int i0 = 0; float p0 = p[0];
#pragma unroll
    for (int e = 1; e < NEXP; ++e) if (p[e] > p0) { p0 = p[e]; i0 = e; }
    int i1 = -1; float p1 = -1.0f;
#pragma unroll
    for (int e = 0; e < NEXP; ++e) if (e != i0 && p[e] > p1) { p1 = p[e]; i1 = e; }

    if (lane == 0) {
        atomicAdd(ent_acc, ent);
        int q0 = atomicAdd(&cnt[i0], 1);
        tok_list[i0 * N_TOK + q0] = n;  prob_list[i0 * N_TOK + q0] = p0;
        int q1 = atomicAdd(&cnt[i1], 1);
        tok_list[i1 * N_TOK + q1] = n;  prob_list[i1 * N_TOK + q1] = p1;
    }
}

// ---------------------------------------------------------------------------
// Aux loss scalar
// ---------------------------------------------------------------------------
__global__ void moe_loss(const int* __restrict__ cnt,
                         const float* __restrict__ ent_acc,
                         float* __restrict__ out_loss)
{
    if (threadIdx.x == 0 && blockIdx.x == 0) {
        float loss = 0.1f * (*ent_acc / (float)N_TOK);
#pragma unroll
        for (int e = 0; e < NEXP; ++e) {
            float ur = (float)cnt[e] / (float)N_TOK;
            loss += fmaxf(ur - 0.3f, 0.0f);
        }
        *out_loss = loss;
    }
}

// ---------------------------------------------------------------------------
// Grouped expert GEMM: 128x128 tile, 8 waves, each wave = 16 rows x 128 cols
// via 8x v_wmma_f32_16x16x32_bf16. fp32->bf16 convert on the fly; fp32 acc.
// LDS double-buffered (1 barrier / k-step), global loads register-pipelined,
// B fragments rotated through a 2-deep pipeline so WMMAs overlap ds loads.
// ---------------------------------------------------------------------------
__global__ __launch_bounds__(256) void moe_expert_gemm(
    const float* __restrict__ x, const float* __restrict__ expert_w,
    const float* __restrict__ expert_b,
    const int* __restrict__ cnt, const int* __restrict__ tok_list,
    const float* __restrict__ prob_list, float* __restrict__ out)
{
    __shared__ __align__(16) unsigned short As[2 * TILE_M * LDSS];
    __shared__ __align__(16) unsigned short Bs[2 * TILE_N * LDSS];
    __shared__ int   toks[TILE_M];
    __shared__ float prbs[TILE_M];

    const int e     = blockIdx.x >> 6;
    const int tm    = blockIdx.x & (MAXTM - 1);
    const int count = cnt[e];
    if (tm * TILE_M >= count) return;
    const int m_base = tm * TILE_M;
    const int n0     = blockIdx.y * TILE_N;

    const int tid  = threadIdx.x;
    const int lane = tid & 31;
    const int wave = tid >> 5;

    if (tid < TILE_M) {
        int r  = m_base + tid;
        int rr = (r < count) ? r : (count - 1);
        toks[tid] = tok_list[e * N_TOK + rr];
        prbs[tid] = (r < count) ? prob_list[e * N_TOK + rr] : 0.0f;
    }
    __syncthreads();

    // staging: each thread owns 16 floats of one row (row = tid/2, half = tid&1)
    const int row = tid >> 1;
    const int seg = tid & 1;
    const float* agp = x        + (long)toks[row] * DDIM + seg * 16;
    const float* bgp = expert_w + ((long)e * HDIM + (n0 + row)) * DDIM + seg * 16;

    v8f acc[8] = {};

    const int hi   = lane >> 4;        // K-halves selector within fragments
    const int ln16 = lane & 15;
    const int mrow = (wave << 4) + ln16;

    float4 ra[4], rb[4];
    auto loadT = [&](int kt) {
        const float* a2 = agp + kt * KSTEP;
        const float* b2 = bgp + kt * KSTEP;
#pragma unroll
        for (int i = 0; i < 4; ++i) {
            ra[i] = ((const float4*)a2)[i];
            rb[i] = ((const float4*)b2)[i];
        }
    };
    auto storeT = [&](int buf) {
        Pack16 pa, pb;
#pragma unroll
        for (int i = 0; i < 4; ++i) {
            pa.b[i * 4 + 0] = (__bf16)ra[i].x; pa.b[i * 4 + 1] = (__bf16)ra[i].y;
            pa.b[i * 4 + 2] = (__bf16)ra[i].z; pa.b[i * 4 + 3] = (__bf16)ra[i].w;
            pb.b[i * 4 + 0] = (__bf16)rb[i].x; pb.b[i * 4 + 1] = (__bf16)rb[i].y;
            pb.b[i * 4 + 2] = (__bf16)rb[i].z; pb.b[i * 4 + 3] = (__bf16)rb[i].w;
        }
        uint4* ap = (uint4*)&As[buf * TILE_M * LDSS + row * LDSS + seg * 16];
        uint4* bp = (uint4*)&Bs[buf * TILE_N * LDSS + row * LDSS + seg * 16];
        ap[0] = pa.q[0]; ap[1] = pa.q[1];
        bp[0] = pb.q[0]; bp[1] = pb.q[1];
    };

    // ---- pipeline prologue: tile 0 -> LDS buf0, tile 1 in flight in regs
    loadT(0);
    storeT(0);
    loadT(1);
    __syncthreads();

    for (int kt = 0; kt < NKITER; ++kt) {
        const int cur = kt & 1;
        const unsigned short* Ab = &As[cur * TILE_M * LDSS];
        const unsigned short* Bb = &Bs[cur * TILE_N * LDSS];

        // A fragment (lanes 0-15: K lo-half, 16-31: K hi-half)
        FragU a;
        a.q[0] = *(const uint4*)(&Ab[mrow * LDSS + hi * 16]);
        a.q[1] = *(const uint4*)(&Ab[mrow * LDSS + hi * 16 + 8]);

        // B fragments: 2-deep rotation so wmma(t) overlaps ds loads of t+1
        FragU bcur, bnxt;
        bcur.q[0] = *(const uint4*)(&Bb[ln16 * LDSS + hi * 16]);
        bcur.q[1] = *(const uint4*)(&Bb[ln16 * LDSS + hi * 16 + 8]);
#pragma unroll
        for (int t = 0; t < 8; ++t) {
            if (t < 7) {
                const int nrow = ((t + 1) << 4) + ln16;
                bnxt.q[0] = *(const uint4*)(&Bb[nrow * LDSS + hi * 16]);
                bnxt.q[1] = *(const uint4*)(&Bb[nrow * LDSS + hi * 16 + 8]);
            }
            acc[t] = __builtin_amdgcn_wmma_f32_16x16x32_bf16(
                         false, a.v, false, bcur.v, (short)0, acc[t], false, false);
            bcur = bnxt;
        }

        if (kt + 1 < NKITER) {
            storeT(cur ^ 1);                 // waits for tile kt+1 loads (in flight during WMMAs)
            if (kt + 2 < NKITER) {
                loadT(kt + 2);               // next tile into regs
                if (kt + 3 < NKITER) {
                    __builtin_prefetch(bgp + (kt + 3) * KSTEP, 0, 1);
                    __builtin_prefetch(agp + (kt + 3) * KSTEP, 0, 1);
                }
            }
            __syncthreads();                 // single barrier per k-step
        }
    }

    // epilogue: out[token, col] += p * (acc + bias)
#pragma unroll
    for (int t = 0; t < 8; ++t) {
        const int col  = n0 + (t << 4) + ln16;
        const float bv = expert_b[e * HDIM + col];
#pragma unroll
        for (int r = 0; r < 8; ++r) {
            const int mrel = (wave << 4) + r + hi * 8;
            if (m_base + mrel < count) {
                const float p = prbs[mrel];
                const float v = p * (acc[t][r] + bv);
                atomicAdd(&out[(long)toks[mrel] * HDIM + col], v);
            }
        }
    }
}

// ---------------------------------------------------------------------------
extern "C" void kernel_launch(void* const* d_in, const int* in_sizes, int n_in,
                              void* d_out, int out_size, void* d_ws, size_t ws_size,
                              hipStream_t stream) {
    const float* x        = (const float*)d_in[0];
    const float* gate_w   = (const float*)d_in[1];
    const float* gate_b   = (const float*)d_in[2];
    const float* expert_w = (const float*)d_in[3];
    const float* expert_b = (const float*)d_in[4];
    float* out = (float*)d_out;

    char*  ws        = (char*)d_ws;
    int*   cnt       = (int*)ws;                       // 8 ints
    float* ent_acc   = (float*)(ws + 32);              // 1 float
    int*   tok_list  = (int*)(ws + 64);                // E*N ints
    float* prob_list = (float*)(ws + 64 + (size_t)NEXP * N_TOK * 4);

    hipMemsetAsync(ws, 0, 64, stream);
    hipMemsetAsync(d_out, 0, (size_t)N_TOK * HDIM * sizeof(float), stream);

    moe_gate<<<N_TOK / 8, 256, 0, stream>>>(x, gate_w, gate_b,
                                            cnt, ent_acc, tok_list, prob_list);
    moe_loss<<<1, 32, 0, stream>>>(cnt, ent_acc, out + (size_t)N_TOK * HDIM);

    dim3 grid(NEXP * MAXTM, HDIM / TILE_N);
    moe_expert_gemm<<<grid, 256, 0, stream>>>(x, expert_w, expert_b,
                                              cnt, tok_list, prob_list, out);
}
